// MMAttention_22548578304495
// MI455X (gfx1250) — compile-verified, matrix operand order checked by reading, not run on hardware
//
#include <hip/hip_runtime.h>

typedef __bf16 bf16;
typedef __attribute__((ext_vector_type(16))) __bf16 v16bf;
typedef __attribute__((ext_vector_type(8)))  __bf16 v8bf;
typedef __attribute__((ext_vector_type(4)))  __bf16 v4bf;
typedef __attribute__((ext_vector_type(2)))  __bf16 v2bf;
typedef __attribute__((ext_vector_type(8)))  float  v8f;

#define DEV_INLINE __device__ __forceinline__

// ---------------------------------------------------------------------------
// WMMA helpers (CDNA5 gfx1250, wave32)
// ---------------------------------------------------------------------------
DEV_INLINE v8f vzero8() {
    v8f v = {0.f, 0.f, 0.f, 0.f, 0.f, 0.f, 0.f, 0.f};
    return v;
}

DEV_INLINE v8f wmma_bf16(v16bf a, v16bf b, v8f c) {
    // D = A(16x32 bf16) x B(32x16 bf16) + C(16x16 f32)
    return __builtin_amdgcn_wmma_f32_16x16x32_bf16(
        /*neg_a=*/false, a, /*neg_b=*/false, b,
        /*c_mod=*/(short)0, c, /*reuse_a=*/false, /*reuse_b=*/false);
}

// Load an A-layout (or B-from-row-major-B^T) 16x32 bf16 fragment.
// Lane L holds row M=L&15; elems 0..7 -> K = kg+0..7, elems 8..15 -> kg+16..23,
// kg = (L>>4)*8  =>  two contiguous 16B loads per lane from a row-major tile.
DEV_INLINE v16bf load_frag(const bf16* base, int ld) {
    const int lane = threadIdx.x & 31;
    const int row  = lane & 15;
    const int kg   = (lane >> 4) << 3;
    const bf16* p  = base + row * ld + kg;
    union { v16bf v; v8bf h[2]; } u;
    u.h[0] = *(const v8bf*)(p);
    u.h[1] = *(const v8bf*)(p + 16);
    return u.v;
}

// ---------------------------------------------------------------------------
// DPP row_ror rotate-butterfly reductions over a 16-lane row (VALU, no LDS
// port pressure; wave32 = two independent 16-lane rows, which exactly matches
// the WMMA C-layout row groups).
// ---------------------------------------------------------------------------
template <int CTRL>
DEV_INLINE float dpp_rotf(float x) {
    int r = __builtin_amdgcn_update_dpp(0, __builtin_bit_cast(int, x),
                                        CTRL, 0xf, 0xf, true);
    return __builtin_bit_cast(float, r);
}
DEV_INLINE float red_max16(float v) {
    v = fmaxf(v, dpp_rotf<0x128>(v));   // row_ror:8
    v = fmaxf(v, dpp_rotf<0x124>(v));   // row_ror:4
    v = fmaxf(v, dpp_rotf<0x122>(v));   // row_ror:2
    v = fmaxf(v, dpp_rotf<0x121>(v));   // row_ror:1
    return v;
}
DEV_INLINE float red_sum16(float v) {
    v += dpp_rotf<0x128>(v);
    v += dpp_rotf<0x124>(v);
    v += dpp_rotf<0x122>(v);
    v += dpp_rotf<0x121>(v);
    return v;
}

// ---------------------------------------------------------------------------
// Kernel 1: f32 -> bf16 conversion (vectorized x4)
// ---------------------------------------------------------------------------
__global__ void k_cvt_bf16(const float* __restrict__ in, bf16* __restrict__ out, int n4) {
    int i = blockIdx.x * blockDim.x + threadIdx.x;
    if (i < n4) {
        float4 v = ((const float4*)in)[i];
        v4bf o;
        o[0] = (bf16)v.x; o[1] = (bf16)v.y; o[2] = (bf16)v.z; o[3] = (bf16)v.w;
        ((v4bf*)out)[i] = o;
    }
}

// ---------------------------------------------------------------------------
// Kernel 2/5: C[m][n] = sum_k A[m][k] * Bw[n][k]  (B given row-major N x K)
// Block tile 64x256, 8 waves (2x4), wave tile 32x64 -> 8 WMMA : 12 ds_load
// per K-step with 4x A-fragment and 2x B-fragment register reuse.
// A-row remap: arow = (m / a_blk) * a_stride + m % a_blk (strided views).
// ---------------------------------------------------------------------------
template <bool OUT_BF16>
__global__ __launch_bounds__(256) void k_gemm_bt(
    const bf16* __restrict__ A, const bf16* __restrict__ Bw,
    const float* __restrict__ bias, void* __restrict__ Cout,
    int M, int N, int K, int a_blk, int a_stride)
{
    __shared__ bf16 As[64 * 32];      // 4 KB
    __shared__ bf16 Bs[256 * 32];     // 16 KB

    const int t    = threadIdx.x;
    const int lane = t & 31;
    const int wid  = t >> 5;
    const int m0   = blockIdx.x * 64;
    const int n0   = blockIdx.y * 256;
    const int wm   = (wid & 1) * 32;   // wave M offset in tile
    const int wn   = (wid >> 1) * 64;  // wave N offset in tile

    v8f acc[2][4];
    #pragma unroll
    for (int i = 0; i < 2; ++i)
        #pragma unroll
        for (int j = 0; j < 4; ++j) acc[i][j] = vzero8();

    // Cooperative-load addressing (16B chunks)
    const int lr = t >> 2;             // 0..63
    const int lc = (t & 3) * 8;        // k element offset within 32
    const int am = m0 + lr;
    const int arow = (am / a_blk) * a_stride + (am % a_blk);
    const bf16* a_src = A + (size_t)arow * K + lc;
    const bf16* b_src[4];
    #pragma unroll
    for (int i = 0; i < 4; ++i)
        b_src[i] = Bw + (size_t)(n0 + lr + 64 * i) * K + lc;

    for (int k0 = 0; k0 < K; k0 += 32) {
        *(v8bf*)(As + lr * 32 + lc) = *(const v8bf*)(a_src + k0);
        #pragma unroll
        for (int i = 0; i < 4; ++i)
            *(v8bf*)(Bs + (lr + 64 * i) * 32 + lc) = *(const v8bf*)(b_src[i] + k0);
        if (k0 + 32 < K) {
            __builtin_prefetch(a_src + k0 + 32, 0, 0);   // global_prefetch_b8
            #pragma unroll
            for (int i = 0; i < 4; ++i)
                __builtin_prefetch(b_src[i] + k0 + 32, 0, 0);
        }
        __syncthreads();

        v16bf a0 = load_frag(As + wm * 32, 32);
        v16bf a1 = load_frag(As + (wm + 16) * 32, 32);
        #pragma unroll
        for (int j = 0; j < 4; ++j) {
            v16bf bj = load_frag(Bs + (wn + 16 * j) * 32, 32);
            acc[0][j] = wmma_bf16(a0, bj, acc[0][j]);
            acc[1][j] = wmma_bf16(a1, bj, acc[1][j]);
        }
        __syncthreads();
    }

    // C/D layout: lane holds column n = lane&15, rows r + 8*(lane>>4)
    const int cn = lane & 15;
    const int cm = (lane >> 4) * 8;
    #pragma unroll
    for (int i = 0; i < 2; ++i) {
        #pragma unroll
        for (int j = 0; j < 4; ++j) {
            const int mb = m0 + wm + i * 16 + cm;
            const int nb = n0 + wn + j * 16 + cn;
            #pragma unroll
            for (int r = 0; r < 8; ++r) {
                float v = acc[i][j][r];
                if constexpr (OUT_BF16) {
                    ((bf16*)Cout)[(size_t)(mb + r) * N + nb] = (bf16)v;
                } else {
                    ((float*)Cout)[(size_t)(mb + r) * N + nb] = v + bias[nb];
                }
            }
        }
    }
}

// ---------------------------------------------------------------------------
// Kernel 3: RMS-norm q,k heads + scatter to (B,H,2048,64) q/k and V^T layout.
// One wave per (stream, b, n, part, head): 64 elems = 2 per lane.
// ---------------------------------------------------------------------------
__global__ __launch_bounds__(256) void k_qkv_norm(
    const bf16* __restrict__ qkv1, const bf16* __restrict__ qkv2,
    const float* __restrict__ g1q, const float* __restrict__ g1k,
    const float* __restrict__ g2q, const float* __restrict__ g2k,
    bf16* __restrict__ q, bf16* __restrict__ k, bf16* __restrict__ vT)
{
    const int lane = threadIdx.x & 31;
    const int w = blockIdx.x * 8 + (threadIdx.x >> 5);

    const int s = w / 196608;           // 4*1024*3*16
    int r = w % 196608;
    const int b = r / 49152;            // 1024*48
    r %= 49152;
    const int n = r / 48;
    r %= 48;
    const int part = r / 16;            // 0=q, 1=k, 2=v
    const int h = r % 16;

    const bf16* qkv = s ? qkv2 : qkv1;
    const int d0 = lane * 2;
    const size_t src = (size_t)(b * 1024 + n) * 3072 + (size_t)part * 1024 + h * 64 + d0;
    v2bf xv = *(const v2bf*)(qkv + src);
    float x0 = (float)xv[0], x1 = (float)xv[1];

    const int bh   = b * 16 + h;
    const int npos = s * 1024 + n;

    if (part == 2) {
        // V stored transposed: vT[bh][d][npos] -> cheap B-fragment loads in PV
        vT[((size_t)bh * 64 + d0) * 2048 + npos]     = xv[0];
        vT[((size_t)bh * 64 + d0 + 1) * 2048 + npos] = xv[1];
        return;
    }

    float ss = x0 * x0 + x1 * x1;
    #pragma unroll
    for (int m = 16; m >= 1; m >>= 1) ss += __shfl_xor(ss, m);
    const float rn = rsqrtf(ss * (1.0f / 64.0f) + 1e-5f);

    const float* g = (part == 0) ? (s ? g2q : g1q) : (s ? g2k : g1k);
    v2bf ov;
    ov[0] = (bf16)(x0 * rn * g[d0]);
    ov[1] = (bf16)(x1 * rn * g[d0 + 1]);
    bf16* dst = (part == 0) ? q : k;
    *(v2bf*)(dst + ((size_t)bh * 2048 + npos) * 64 + d0) = ov;
}

// ---------------------------------------------------------------------------
// Kernel 4: flash attention. Grid (B*H=64, 2048/128 qtiles). 128 thr = 4 waves.
// Each wave: 32 query rows (2 row groups sharing K/V fragments -> 16 WMMA per
// 32-key block). Online softmax in exp2 space, DPP row reductions.
// ---------------------------------------------------------------------------
__global__ __launch_bounds__(128) void k_attention(
    const bf16* __restrict__ q, const bf16* __restrict__ k,
    const bf16* __restrict__ vT, bf16* __restrict__ attn_out)
{
    __shared__ bf16 Ks[32 * 64];        // keys x d           (4 KB)
    __shared__ bf16 Vs[64 * 32];        // d x keys (V^T)     (4 KB)
    __shared__ bf16 Ps[4][32 * 32];     // per-wave P scratch (8 KB)

    const int t    = threadIdx.x;
    const int lane = t & 31;
    const int wid  = t >> 5;
    const int bh   = blockIdx.x;
    const int q0   = blockIdx.y * 128 + wid * 32;

    const bf16* qbase = q + ((size_t)bh * 2048 + q0) * 64;
    v16bf qf[2][2];                     // [row group][d slice]
    #pragma unroll
    for (int g = 0; g < 2; ++g) {
        qf[g][0] = load_frag(qbase + g * 16 * 64, 64);
        qf[g][1] = load_frag(qbase + g * 16 * 64 + 32, 64);
    }

    const float sm_scale = 0.125f * 1.4426950408889634f;  // 1/sqrt(64) * log2(e)

    float mrow[2][8], lrow[2][8];
    v8f o[2][4];
    #pragma unroll
    for (int g = 0; g < 2; ++g) {
        #pragma unroll
        for (int i = 0; i < 8; ++i) { mrow[g][i] = -1e30f; lrow[g][i] = 0.f; }
        #pragma unroll
        for (int c = 0; c < 4; ++c) o[g][c] = vzero8();
    }

    for (int kb = 0; kb < 2048; kb += 32) {
        // Cooperative load of K block (32x64) and V^T block (64x32), 16B chunks
        #pragma unroll
        for (int i = 0; i < 2; ++i) {
            int idx = t + i * 128;
            int rk = idx >> 3, ck = (idx & 7) * 8;
            *(v8bf*)(Ks + rk * 64 + ck) =
                *(const v8bf*)(k + ((size_t)bh * 2048 + kb + rk) * 64 + ck);
            int rv = idx >> 2, cv = (idx & 3) * 8;
            *(v8bf*)(Vs + rv * 32 + cv) =
                *(const v8bf*)(vT + ((size_t)bh * 64 + rv) * 2048 + kb + cv);
        }
        __syncthreads();

        // Shared fragments for both row groups
        v16bf kf[2][2];                 // [key col group][d slice]
        kf[0][0] = load_frag(Ks, 64);
        kf[0][1] = load_frag(Ks + 32, 64);
        kf[1][0] = load_frag(Ks + 16 * 64, 64);
        kf[1][1] = load_frag(Ks + 16 * 64 + 32, 64);
        v16bf vf[4];
        #pragma unroll
        for (int c = 0; c < 4; ++c) vf[c] = load_frag(Vs + (c * 16) * 32, 32);

        const int n  = lane & 15;
        const int mh = (lane >> 4) * 8;

        #pragma unroll
        for (int g = 0; g < 2; ++g) {
            // S = Q K^T : 16 rows x 32 keys
            v8f s0 = vzero8(), s1 = vzero8();
            s0 = wmma_bf16(qf[g][0], kf[0][0], s0);
            s0 = wmma_bf16(qf[g][1], kf[0][1], s0);
            s1 = wmma_bf16(qf[g][0], kf[1][0], s1);
            s1 = wmma_bf16(qf[g][1], kf[1][1], s1);

            // Online softmax (DPP row_ror butterfly stays within 16-lane row)
            float p0[8], p1[8], alpha[8];
            #pragma unroll
            for (int r = 0; r < 8; ++r) {
                float a  = s0[r] * sm_scale;
                float bb = s1[r] * sm_scale;
                float mx = red_max16(fmaxf(a, bb));
                float mnew = fmaxf(mrow[g][r], mx);
                alpha[r]   = exp2f(mrow[g][r] - mnew);
                mrow[g][r] = mnew;
                p0[r] = exp2f(a - mnew);
                p1[r] = exp2f(bb - mnew);
                float rs = red_sum16(p0[r] + p1[r]);
                lrow[g][r] = lrow[g][r] * alpha[r] + rs;
            }

            // C-layout -> A-layout via wave-private LDS (per-wave DS in order)
            bf16* pw = &Ps[wid][g * 16 * 32];
            #pragma unroll
            for (int r = 0; r < 8; ++r) {
                pw[(mh + r) * 32 + n]      = (bf16)p0[r];
                pw[(mh + r) * 32 + n + 16] = (bf16)p1[r];
            }

            // Rescale O accumulators
            #pragma unroll
            for (int c = 0; c < 4; ++c)
                #pragma unroll
                for (int r = 0; r < 8; ++r) o[g][c][r] *= alpha[r];
        }

        // O += P @ V  (A = P 16x32, B chunks from V^T rows; vf reused 2x)
        #pragma unroll
        for (int g = 0; g < 2; ++g) {
            v16bf pf = load_frag(&Ps[wid][g * 16 * 32], 32);
            #pragma unroll
            for (int c = 0; c < 4; ++c)
                o[g][c] = wmma_bf16(pf, vf[c], o[g][c]);
        }
        __syncthreads();
    }

    // Epilogue: O / l -> attn_out (B, npos, h*64+d) as bf16
    const int n  = lane & 15;
    const int mh = (lane >> 4) * 8;
    const int b  = bh >> 4, h = bh & 15;
    #pragma unroll
    for (int g = 0; g < 2; ++g) {
        #pragma unroll
        for (int r = 0; r < 8; ++r) {
            const float inv  = 1.0f / lrow[g][r];
            const int npos   = q0 + g * 16 + mh + r;
            const size_t row = ((size_t)b * 2048 + npos) * 1024 + h * 64;
            #pragma unroll
            for (int c = 0; c < 4; ++c)
                attn_out[row + c * 16 + n] = (bf16)(o[g][c][r] * inv);
        }
    }
}

// ---------------------------------------------------------------------------
// Host launcher
// ---------------------------------------------------------------------------
extern "C" void kernel_launch(void* const* d_in, const int* in_sizes, int n_in,
                              void* d_out, int out_size, void* d_ws, size_t ws_size,
                              hipStream_t stream) {
    (void)in_sizes; (void)n_in; (void)out_size; (void)ws_size;
    const float* x1      = (const float*)d_in[0];
    const float* x2      = (const float*)d_in[1];
    const float* w1_qkv  = (const float*)d_in[2];
    const float* w2_qkv  = (const float*)d_in[3];
    const float* g1_q    = (const float*)d_in[4];
    const float* g1_k    = (const float*)d_in[5];
    const float* g2_q    = (const float*)d_in[6];
    const float* g2_k    = (const float*)d_in[7];
    const float* w1_proj = (const float*)d_in[8];
    const float* b1_proj = (const float*)d_in[9];
    const float* w2_proj = (const float*)d_in[10];
    const float* b2_proj = (const float*)d_in[11];

    constexpr size_t XE   = 4ull * 1024 * 1024;   // x elements
    constexpr size_t WQE  = 3072ull * 1024;
    constexpr size_t WPE  = 1024ull * 1024;
    constexpr size_t QKVE = 4096ull * 3072;
    constexpr size_t HE   = 64ull * 2048 * 64;    // (B*H, 2048, 64)
    constexpr size_t AOE  = 4ull * 2048 * 1024;

    char* ws = (char*)d_ws;
    size_t off = 0;
    auto alloc = [&](size_t elems) {
        char* p = ws + off;
        off += (elems * sizeof(bf16) + 255) & ~size_t(255);
        return (bf16*)p;
    };
    bf16* xb1   = alloc(XE);
    bf16* xb2   = alloc(XE);
    bf16* wq1b  = alloc(WQE);
    bf16* wq2b  = alloc(WQE);
    bf16* wp1b  = alloc(WPE);
    bf16* wp2b  = alloc(WPE);
    bf16* qkv1b = alloc(QKVE);
    bf16* qkv2b = alloc(QKVE);
    bf16* qh    = alloc(HE);
    bf16* kh    = alloc(HE);
    bf16* vTh   = alloc(HE);
    bf16* attn  = alloc(AOE);

    auto cvt = [&](const float* src, bf16* dst, size_t n) {
        int n4 = (int)(n / 4);
        k_cvt_bf16<<<(n4 + 255) / 256, 256, 0, stream>>>(src, dst, n4);
    };
    cvt(x1, xb1, XE);
    cvt(x2, xb2, XE);
    cvt(w1_qkv, wq1b, WQE);
    cvt(w2_qkv, wq2b, WQE);
    cvt(w1_proj, wp1b, WPE);
    cvt(w2_proj, wp2b, WPE);

    // QKV GEMMs: (4096 x 1024) x (1024 x 3072)^T
    dim3 gq(4096 / 64, 3072 / 256);
    k_gemm_bt<true><<<gq, 256, 0, stream>>>(xb1, wq1b, nullptr, qkv1b,
                                            4096, 3072, 1024, 4096, 4096);
    k_gemm_bt<true><<<gq, 256, 0, stream>>>(xb2, wq2b, nullptr, qkv2b,
                                            4096, 3072, 1024, 4096, 4096);

    // RMS-norm + head reshape + V transpose
    k_qkv_norm<<<49152, 256, 0, stream>>>(qkv1b, qkv2b, g1_q, g1_k, g2_q, g2_k,
                                          qh, kh, vTh);

    // Flash attention over N=2048
    dim3 ga(64, 16);
    k_attention<<<ga, 128, 0, stream>>>(qh, kh, vTh, attn);

    // Projections: stream rows are strided views of attn (B, 2048, 1024)
    dim3 gp(4096 / 64, 1024 / 256);
    float* out1 = (float*)d_out;
    float* out2 = out1 + 4ull * 1024 * 1024;
    k_gemm_bt<false><<<gp, 256, 0, stream>>>(attn, wp1b, b1_proj, out1,
                                             4096, 1024, 1024, 1024, 2048);
    k_gemm_bt<false><<<gp, 256, 0, stream>>>(attn + 1024 * 1024, wp2b, b2_proj, out2,
                                             4096, 1024, 1024, 1024, 2048);
}
